// LinkLevelGNN_429496730105
// MI455X (gfx1250) — matrix operand order repaired and checked by reading, not compiled, not run on hardware
//
#include <hip/hip_runtime.h>
#include <hip/hip_bf16.h>
#include <stdint.h>

// ---------------------------------------------------------------------------
// Types for CDNA5 WMMA
// ---------------------------------------------------------------------------
typedef __attribute__((ext_vector_type(16))) __bf16 v16bf;
typedef __attribute__((ext_vector_type(8)))  float  v8f;

struct bf16x8 { unsigned short u[8]; };   // 16-byte chunk of 8 bf16 values

__device__ __forceinline__ unsigned short f32_to_bf16_bits(float f) {
    // round-to-nearest-even bf16
    uint32_t x = __builtin_bit_cast(uint32_t, f);
    uint32_t lsb = (x >> 16) & 1u;
    x += 0x7FFFu + lsb;
    return (unsigned short)(x >> 16);
}

// ---------------------------------------------------------------------------
// Elementwise / prep kernels
// ---------------------------------------------------------------------------
__global__ void cast_f32_to_bf16_kernel(const float* __restrict__ in,
                                        unsigned short* __restrict__ out, int n) {
    int i = blockIdx.x * blockDim.x + threadIdx.x;
    if (i < n) out[i] = f32_to_bf16_bits(in[i]);
}

// Wt[n*K + k] = bf16(W[k*N + n])   (W is [K x N] row-major)
__global__ void transpose_cast_kernel(const float* __restrict__ W,
                                      unsigned short* __restrict__ Wt,
                                      int K, int N) {
    int i = blockIdx.x * blockDim.x + threadIdx.x;
    if (i >= K * N) return;
    int n = i / K;
    int k = i - n * K;
    Wt[i] = f32_to_bf16_bits(W[(size_t)k * N + n]);
}

__global__ void fill_f32_kernel(float* __restrict__ p, float v, int n) {
    int i = blockIdx.x * blockDim.x + threadIdx.x;
    if (i < n) p[i] = v;
}

__global__ void deg_accum_kernel(const long long* __restrict__ dst,
                                 float* __restrict__ deg, long long E) {
    long long i = (long long)blockIdx.x * blockDim.x + threadIdx.x;
    if (i < E) atomicAdd(&deg[dst[i]], 1.0f);
}

__global__ void rsqrt_inplace_kernel(float* __restrict__ d, int n) {
    int i = blockIdx.x * blockDim.x + threadIdx.x;
    if (i < n) d[i] = rsqrtf(d[i]);   // deg >= 1 always (self-loop)
}

// ---------------------------------------------------------------------------
// bf16 WMMA GEMM:  C[M x N] = A[M x K] @ Bt[N x K]^T  (fp32 accumulate)
// One wave per 16x16 tile; 8 waves per block cover N=128.
// Fragment layouts used by this kernel (wave32; k = 16h + s, same in A and B):
//   A 16x32 bf16: lane(l): row = l&15, half h = l>>4;
//     elems 0..7  = A[row][k0 + 16h .. k0+16h+7]      (16B contiguous)
//     elems 8..15 = A[row][k0+16h+8 .. k0+16h+15]     (16B contiguous)
//   B 32x16 bf16 (from transposed weights): lane(l): col = l&15;
//     elems 0..15 = Bt[col][k0 + 16h .. k0+16h+15]    (32B contiguous)
//   C/D 16x16 f32: lane(l): col = l&15; vgpr j -> row = 8h + j
// ---------------------------------------------------------------------------
__global__ __launch_bounds__(256)
void wmma_gemm_bf16_kernel(const unsigned short* __restrict__ A,
                           const unsigned short* __restrict__ Bt,
                           float* __restrict__ C,
                           int M, int N, int K) {
    const int wave = threadIdx.x >> 5;
    const int lane = threadIdx.x & 31;
    const int mBase = blockIdx.x * 16;
    const int nBase = wave * 16;
    if (mBase >= M || nBase >= N) return;

    const int r = lane & 15;
    const int hh = lane >> 4;

    const unsigned short* arow = A  + (size_t)(mBase + r) * K;
    const unsigned short* brow = Bt + (size_t)(nBase + r) * K;

    v8f acc = {};
    for (int k0 = 0; k0 < K; k0 += 32) {
        bf16x8 a0 = *(const bf16x8*)(arow + k0 + 16 * hh);
        bf16x8 a1 = *(const bf16x8*)(arow + k0 + 16 * hh + 8);
        v16bf af;
#pragma unroll
        for (int i = 0; i < 8; ++i) {
            af[i]     = __builtin_bit_cast(__bf16, a0.u[i]);
            af[i + 8] = __builtin_bit_cast(__bf16, a1.u[i]);
        }
        v16bf bf = *(const v16bf*)(brow + k0 + 16 * hh);

        acc = __builtin_amdgcn_wmma_f32_16x16x32_bf16(
            /*neg_a=*/false, af, /*neg_b=*/false, bf,
            /*c_mod=*/(short)0, acc, /*reuse_a=*/false, /*reuse_b=*/false);
    }

    float* cptr = C + (size_t)(mBase + 8 * hh) * N + nBase + r;
#pragma unroll
    for (int j = 0; j < 8; ++j) cptr[(size_t)j * N] = acc[j];
}

// ---------------------------------------------------------------------------
// Edge aggregation: one wave (32 lanes) per edge, 128 channels = float4/lane.
//   acc[dst] += dinv[src]*dinv[dst] * h[src]
// ---------------------------------------------------------------------------
__global__ __launch_bounds__(256)
void edge_agg_kernel(const float* __restrict__ h,
                     const long long* __restrict__ src,
                     const long long* __restrict__ dst,
                     const float* __restrict__ dinv,
                     float* __restrict__ acc, long long E) {
    long long e = ((long long)blockIdx.x * blockDim.x + threadIdx.x) >> 5;
    int lane = threadIdx.x & 31;
    if (e >= E) return;
    long long s = src[e];
    long long d = dst[e];
    float w = dinv[s] * dinv[d];
    float4 v = ((const float4*)(h + s * 128))[lane];
    float* ap = acc + d * 128 + lane * 4;
    atomicAdd(ap + 0, w * v.x);
    atomicAdd(ap + 1, w * v.y);
    atomicAdd(ap + 2, w * v.z);
    atomicAdd(ap + 3, w * v.w);
}

// Epilogue layer 1: h2 = relu(acc + dinv^2 * h1 + b1), cast to bf16 for GEMM2.
__global__ void epilogue1_kernel(const float* __restrict__ acc,
                                 const float* __restrict__ h1,
                                 const float* __restrict__ dinv,
                                 const float* __restrict__ b,
                                 unsigned short* __restrict__ h2b,
                                 int Nn) {
    int i = blockIdx.x * blockDim.x + threadIdx.x;
    if (i >= Nn * 128) return;
    int row = i >> 7;
    int col = i & 127;
    float di = dinv[row];
    float v = acc[i] + di * di * h1[i] + b[col];
    v = fmaxf(v, 0.0f);
    h2b[i] = f32_to_bf16_bits(v);
}

// Epilogue layer 2: out = acc + dinv^2 * g2 + b2  (fp32 output)
__global__ void epilogue2_kernel(const float* __restrict__ acc,
                                 const float* __restrict__ g2,
                                 const float* __restrict__ dinv,
                                 const float* __restrict__ b,
                                 float* __restrict__ out,
                                 int Nn) {
    int i = blockIdx.x * blockDim.x + threadIdx.x;
    if (i >= Nn * 128) return;
    int row = i >> 7;
    int col = i & 127;
    float di = dinv[row];
    out[i] = acc[i] + di * di * g2[i] + b[col];
}

// ---------------------------------------------------------------------------
// Launcher
// ---------------------------------------------------------------------------
extern "C" void kernel_launch(void* const* d_in, const int* in_sizes, int n_in,
                              void* d_out, int out_size, void* d_ws, size_t ws_size,
                              hipStream_t stream) {
    (void)n_in; (void)out_size; (void)ws_size;

    const float*     x    = (const float*)d_in[0];
    const long long* eidx = (const long long*)d_in[1];   // int64 [2, E] flat
    const float*     W1   = (const float*)d_in[2];
    const float*     b1   = (const float*)d_in[3];
    const float*     W2   = (const float*)d_in[4];
    const float*     b2   = (const float*)d_in[5];
    float*           out  = (float*)d_out;

    const int IN = 384, H = 128;
    const int Nn = in_sizes[0] / IN;                     // 100000
    const long long E = (long long)in_sizes[1] / 2;      // 3200000
    const long long* src = eidx;
    const long long* dst = eidx + E;

    // ---- workspace carve (256B aligned) ----
    char* p = (char*)d_ws;
    auto carve = [&](size_t bytes) {
        char* q = p;
        p += (bytes + 255) & ~(size_t)255;
        return q;
    };
    unsigned short* xb   = (unsigned short*)carve((size_t)Nn * IN * 2);  // x in bf16
    unsigned short* w1t  = (unsigned short*)carve((size_t)IN * H * 2);   // W1^T bf16 [H x IN]
    unsigned short* w2t  = (unsigned short*)carve((size_t)H * H * 2);    // W2^T bf16 [H x H]
    float*          dinv = (float*)carve((size_t)Nn * 4);                // deg -> rsqrt in place
    float*          hbuf = (float*)carve((size_t)Nn * H * 4);            // h1, reused for g2
    float*          abuf = (float*)carve((size_t)Nn * H * 4);            // agg accum (both layers)
    unsigned short* h2b  = (unsigned short*)carve((size_t)Nn * H * 2);   // relu output in bf16

    const int T = 256;
    auto cdiv = [](long long a, long long b) { return (int)((a + b - 1) / b); };

    // 1) casts / transposes
    cast_f32_to_bf16_kernel<<<cdiv((long long)Nn * IN, T), T, 0, stream>>>(x, xb, Nn * IN);
    transpose_cast_kernel<<<cdiv((long long)IN * H, T), T, 0, stream>>>(W1, w1t, IN, H);
    transpose_cast_kernel<<<cdiv((long long)H * H, T), T, 0, stream>>>(W2, w2t, H, H);

    // 2) degrees (self-loop => init 1.0), then dinv = rsqrt(deg)
    fill_f32_kernel<<<cdiv(Nn, T), T, 0, stream>>>(dinv, 1.0f, Nn);
    deg_accum_kernel<<<cdiv(E, T), T, 0, stream>>>(dst, dinv, E);
    rsqrt_inplace_kernel<<<cdiv(Nn, T), T, 0, stream>>>(dinv, Nn);

    // 3) GEMM1: h1 = xb @ W1   (M=Nn, N=H, K=IN)  -- 8 waves/block cover N=128
    wmma_gemm_bf16_kernel<<<Nn / 16, 256, 0, stream>>>(xb, w1t, hbuf, Nn, H, IN);

    // 4) aggregate layer 1
    fill_f32_kernel<<<cdiv((long long)Nn * H, T), T, 0, stream>>>(abuf, 0.0f, Nn * H);
    edge_agg_kernel<<<cdiv(E * 32, T), T, 0, stream>>>(hbuf, src, dst, dinv, abuf, E);
    epilogue1_kernel<<<cdiv((long long)Nn * H, T), T, 0, stream>>>(abuf, hbuf, dinv, b1, h2b, Nn);

    // 5) GEMM2: g2 = h2 @ W2   (M=Nn, N=H, K=H), reuse hbuf for g2
    wmma_gemm_bf16_kernel<<<Nn / 16, 256, 0, stream>>>(h2b, w2t, hbuf, Nn, H, H);

    // 6) aggregate layer 2 + final epilogue
    fill_f32_kernel<<<cdiv((long long)Nn * H, T), T, 0, stream>>>(abuf, 0.0f, Nn * H);
    edge_agg_kernel<<<cdiv(E * 32, T), T, 0, stream>>>(hbuf, src, dst, dinv, abuf, E);
    epilogue2_kernel<<<cdiv((long long)Nn * H, T), T, 0, stream>>>(abuf, hbuf, dinv, b2, out, Nn);
}
